// MemEffAttn_1838246003298
// MI455X (gfx1250) — matrix operand order repaired and checked by reading, not compiled, not run on hardware
//
#include <hip/hip_runtime.h>

#define T_SEQ 1024
#define BATCH 4
#define EMB   1024
#define HEADS 16
#define HD    64
#define NROW  (T_SEQ * BATCH)   // 4096

typedef __attribute__((ext_vector_type(16))) __bf16 v16bf;
typedef __attribute__((ext_vector_type(8)))  float  v8f;
typedef unsigned int u32x4 __attribute__((ext_vector_type(4)));
typedef int          i32x4 __attribute__((ext_vector_type(4)));
typedef int          i32x8 __attribute__((ext_vector_type(8)));

union BFr { unsigned int u[8]; v16bf v; };

__device__ inline unsigned short f2bf(float f) {
  // round-to-nearest-even f32 -> bf16
  unsigned int x = __builtin_bit_cast(unsigned int, f);
  unsigned int r = x + 0x7fffu + ((x >> 16) & 1u);
  return (unsigned short)(r >> 16);
}

__device__ inline v8f wmma_bf16(const BFr& a, const BFr& b, v8f c) {
  return __builtin_amdgcn_wmma_f32_16x16x32_bf16(
      false, a.v, false, b.v, (short)0, c, false, false);
}

// ---------------------------------------------------------------------------
// Kernel 1: fused QKV projection + RoPE + head reshape.
//   out[n, j] = sum_e X[n,e] * W[j,e] (+bias)  ->  RoPE(q,k) -> bf16 [B][H][T][D]
// block = 128 (4 waves); each block computes a 64x64 tile; grid.z picks q/k/v.
// ---------------------------------------------------------------------------
__global__ __launch_bounds__(128) void qkv_proj_rope_kernel(
    const float* __restrict__ X,
    const float* __restrict__ Wq, const float* __restrict__ bq,
    const float* __restrict__ Wk,
    const float* __restrict__ Wv, const float* __restrict__ bv,
    unsigned short* __restrict__ Qh, unsigned short* __restrict__ Kh,
    unsigned short* __restrict__ Vh)
{
  __shared__ unsigned short Xs[64 * 32];
  __shared__ unsigned short Ws[64 * 32];
  __shared__ float          Es[64 * 64];

  const int tid  = threadIdx.x;
  const int lane = tid & 31;
  const int w    = tid >> 5;
  const int hg   = lane >> 4;   // half of wave (0/1)
  const int ln   = lane & 15;

  const int n0 = blockIdx.x * 64;           // row tile in N = T*B
  const int h  = blockIdx.y;                // 64-col tile == head index
  const int z  = blockIdx.z;                // 0=q, 1=k, 2=v

  const float* W      = (z == 0) ? Wq : ((z == 1) ? Wk : Wv);
  const float* bias   = (z == 2) ? bv : bq;        // always valid pointer
  const float  bscale = (z == 1) ? 0.f : 1.f;      // k-proj has no bias
  unsigned short* dst = (z == 0) ? Qh : ((z == 1) ? Kh : Vh);

  v8f acc[4];
#pragma unroll
  for (int cg = 0; cg < 4; ++cg) acc[cg] = (v8f){0,0,0,0,0,0,0,0};

  for (int kk = 0; kk < EMB; kk += 32) {
    // stage X (64x32) and W (64x32) tiles as bf16
    for (int idx = tid; idx < 64 * 32; idx += 128) {
      int r = idx >> 5, c = idx & 31;
      Xs[idx] = f2bf(X[(size_t)(n0 + r) * EMB + kk + c]);
      Ws[idx] = f2bf(W[(size_t)(h * 64 + r) * EMB + kk + c]);
    }
    __syncthreads();

    const unsigned int* xs32 = (const unsigned int*)Xs;  // 16 dwords / row
    const unsigned int* ws32 = (const unsigned int*)Ws;

    BFr a;  // A: 16 rows (this wave) x 32 k
    {
      int base = (16 * w + ln) * 16 + 8 * hg;
#pragma unroll
      for (int v = 0; v < 4; ++v) {
        a.u[v]     = xs32[base + v];
        a.u[4 + v] = xs32[base + 4 + v];
      }
    }
#pragma unroll
    for (int cg = 0; cg < 4; ++cg) {
      BFr bfr;  // B: 32 k x 16 out-cols; lane = col, k = 16*hg + 2v
      int jb = (16 * cg + ln) * 16 + 8 * hg;
#pragma unroll
      for (int v = 0; v < 8; ++v) bfr.u[v] = ws32[jb + v];
      acc[cg] = wmma_bf16(a, bfr, acc[cg]);
    }
    __syncthreads();
  }

  // bias + spill to LDS f32 (RoPE loop then pairs d with d^32 of the same row)
#pragma unroll
  for (int cg = 0; cg < 4; ++cg) {
    int col = 16 * cg + ln;
    float bb = bias[h * 64 + col] * bscale;
#pragma unroll
    for (int g = 0; g < 8; ++g)
      Es[(16 * w + g + 8 * hg) * 64 + col] = acc[cg][g] + bb;
  }
  __syncthreads();

  // RoPE (q,k only) + store bf16 in [B][H][T][D]
  for (int idx = tid; idx < 64 * 64; idx += 128) {
    int r = idx >> 6, d = idx & 63;
    int n = n0 + r, t = n >> 2, b = n & 3;   // n = t*B + b, B = 4
    float val = Es[idx];
    if (z < 2) {
      float partner = Es[idx ^ 32];          // same row, d^32
      int   j   = d & 31;
      // inv_freq = 10000^(-j/32) = exp(-j * ln(10000)/32)
      float inv = expf(-(float)j * 0.28782313662425572f);
      float ang = (float)t * inv;
      float sn, cs;
      sincosf(ang, &sn, &cs);
      float rot = (d < 32) ? -partner : partner;
      val = val * cs + rot * sn;
    }
    dst[((size_t)(b * HEADS + h) * T_SEQ + t) * HD + d] = f2bf(val);
  }
}

// ---------------------------------------------------------------------------
// Kernel 2: flash attention. block = 128 (4 waves), each wave owns 16 q-rows.
// grid = (T/64, H, B). K tile staged by the Tensor Data Mover (TDM) with
// hardware row padding (1 dword every 32 dwords -> 66-bf16 rows); V staged
// transposed manually; attn_bias (256 MB) streamed with non-temporal loads.
// ---------------------------------------------------------------------------
__global__ __launch_bounds__(128) void flash_attn_kernel(
    const unsigned short* __restrict__ Qh, const unsigned short* __restrict__ Kh,
    const unsigned short* __restrict__ Vh, const float* __restrict__ attn_bias,
    const unsigned char* __restrict__ kpm, float* __restrict__ O)
{
  __shared__ unsigned short Ks[64 * 66];        // K block, row-major [s][d], padded
  __shared__ unsigned short VT[64 * 66];        // V block, transposed [d][s], padded
  __shared__ unsigned short Ps[4 * 16 * 66];    // per-wave P tile [m][s], padded

  const int tid  = threadIdx.x;
  const int lane = tid & 31;
  const int w    = tid >> 5;
  const int hg   = lane >> 4;
  const int ln   = lane & 15;

  const int t0 = blockIdx.x * 64;
  const int h  = blockIdx.y;
  const int b  = blockIdx.z;
  const int bh = b * HEADS + h;
  const size_t hbase = (size_t)bh * T_SEQ * HD;   // bf16 elements

  // LDS byte offset of Ks: generic LDS addr low 32 bits == LDS offset (ISA 10.2)
  const unsigned ks_lds = (unsigned)(unsigned long long)(const void*)Ks;

  // Load this wave's Q A-fragments (16 rows x 64 d => 2 k-steps) from global
  BFr qf[2];
  {
    int trow = t0 + 16 * w + ln;
    const unsigned int* qp = (const unsigned int*)(Qh + hbase + (size_t)trow * HD);
#pragma unroll
    for (int ks = 0; ks < 2; ++ks)
#pragma unroll
      for (int v = 0; v < 8; ++v) {
        int k = 2 * v + 16 * hg + 32 * ks;  // even
        qf[ks].u[v] = qp[k >> 1];
      }
  }

  float m_run[8], l_run[8];
#pragma unroll
  for (int g = 0; g < 8; ++g) { m_run[g] = -INFINITY; l_run[g] = 0.f; }
  v8f accO[4];
#pragma unroll
  for (int cg = 0; cg < 4; ++cg) accO[cg] = (v8f){0,0,0,0,0,0,0,0};

  const float scale = 0.125f;   // 1/sqrt(64)
  const int pwbase = w * 16 * 66;

  for (int s0 = 0; s0 < T_SEQ; s0 += 64) {
    // ---- K tile via Tensor Data Mover (wave 0 issues, TENSORcnt sync) ----
    if (w == 0) {
      unsigned long long ga =
          (unsigned long long)(const void*)(Kh + hbase + (size_t)s0 * HD);
      u32x4 g0;
      g0[0] = 1u;                                   // count=1, user descriptor
      g0[1] = ks_lds;                               // lds_addr (bytes)
      g0[2] = (unsigned)(ga & 0xffffffffull);       // global_addr[31:0]
      g0[3] = (unsigned)((ga >> 32) & 0x01ffffffull) | (2u << 30);  // [56:32] | type=2
      i32x8 g1;
      g1[0] = (int)((1u << 16)    // data_size = 1 -> 2-byte elements
                  | (1u << 20)    // pad_enable
                  | (4u << 22));  // pad_interval = 32 dwords; pad_amount=0 -> 1 dword
      g1[1] = (int)(64u << 16);   // tensor_dim0 = 64 (d)
      g1[2] = (int)(64u << 16);   // tensor_dim1 = 64 (s)  [dim0 hi bits = 0]
      g1[3] = (int)(64u << 16);   // tile_dim0 = 64        [dim1 hi bits = 0]
      g1[4] = 64;                 // tile_dim1 = 64, tile_dim2 = 0
      g1[5] = 64;                 // tensor_dim0_stride = 64 (row stride in elems)
      g1[6] = 0;                  // dim0_stride hi = 0, tensor_dim1_stride lo = 0
      g1[7] = 0;
      i32x4 gz = {0, 0, 0, 0};    // 2-D tile: groups 2/3 unused
#if defined(__clang_major__) && (__clang_major__ >= 23)
      i32x8 gz8 = {0, 0, 0, 0, 0, 0, 0, 0};
      __builtin_amdgcn_tensor_load_to_lds(g0, g1, gz, gz, gz8, 0);
#else
      __builtin_amdgcn_tensor_load_to_lds(g0, g1, gz, gz, 0);
#endif
      __builtin_amdgcn_s_wait_tensorcnt(0);
    }

    // ---- V tile staged transposed: VT[d][s] ----
    for (int idx = tid; idx < 64 * 32; idx += 128) {
      int s = idx >> 5, dp = idx & 31;  // dword pair index (2 bf16)
      unsigned int vw = *(const unsigned int*)(Vh + hbase + (size_t)(s0 + s) * HD + 2 * dp);
      VT[(2 * dp)     * 66 + s] = (unsigned short)(vw & 0xffffu);
      VT[(2 * dp + 1) * 66 + s] = (unsigned short)(vw >> 16);
    }
    __syncthreads();

    // S = (Q K^T) * scale + bias + mask   (4 col-groups of 16)
    float S[4][8];
    const unsigned int* ks32 = (const unsigned int*)Ks;   // 33 dwords / row
#pragma unroll
    for (int cg = 0; cg < 4; ++cg) {
      v8f c = (v8f){0,0,0,0,0,0,0,0};
      int srow = 16 * cg + ln;
#pragma unroll
      for (int ks = 0; ks < 2; ++ks) {
        BFr bf;
        int base = srow * 33 + 8 * hg + 16 * ks;
#pragma unroll
        for (int v = 0; v < 8; ++v) bf.u[v] = ks32[base + v];
        c = wmma_bf16(qf[ks], bf, c);
      }
      int   scol = s0 + 16 * cg + ln;
      float mval = kpm[b * T_SEQ + scol] ? -INFINITY : 0.f;
      size_t brow0 = ((size_t)bh * T_SEQ + (t0 + 16 * w + 8 * hg)) * T_SEQ + scol;
#pragma unroll
      for (int g = 0; g < 8; ++g) {
        float bias_v = __builtin_nontemporal_load(attn_bias + brow0 + (size_t)g * T_SEQ);
        S[cg][g] = c[g] * scale + bias_v + mval;
      }
    }

    // online softmax per row (rows replicated across 16 lanes of each half)
#pragma unroll
    for (int g = 0; g < 8; ++g) {
      float mx = fmaxf(fmaxf(S[0][g], S[1][g]), fmaxf(S[2][g], S[3][g]));
#pragma unroll
      for (int off = 1; off < 16; off <<= 1) mx = fmaxf(mx, __shfl_xor(mx, off, 32));
      float mnew  = fmaxf(m_run[g], mx);
      float msafe = (mnew == -INFINITY) ? 0.f : mnew;
      float fac   = __expf(m_run[g] - msafe);   // -inf -> 0
      float rsum  = 0.f;
      int   m     = g + 8 * hg;
#pragma unroll
      for (int cg = 0; cg < 4; ++cg) {
        float p = __expf(S[cg][g] - msafe);
        rsum += p;
        Ps[pwbase + m * 66 + 16 * cg + ln] = f2bf(p);
        accO[cg][g] = accO[cg][g] * fac;
      }
#pragma unroll
      for (int off = 1; off < 16; off <<= 1) rsum += __shfl_xor(rsum, off, 32);
      l_run[g] = l_run[g] * fac + rsum;
      m_run[g] = mnew;
    }

    // same-wave LDS store->load ordering (HW keeps DS in order; wait for data)
    asm volatile("s_wait_dscnt 0x0" ::: "memory");

    // O += P * V
    const unsigned int* pp   = (const unsigned int*)(Ps + pwbase);  // 33 dw/row
    const unsigned int* vt32 = (const unsigned int*)VT;             // 33 dw/row
#pragma unroll
    for (int cg = 0; cg < 4; ++cg) {
#pragma unroll
      for (int ks = 0; ks < 2; ++ks) {
        BFr af, bf;
        int abase = ln * 33 + 8 * hg + 16 * ks;
#pragma unroll
        for (int v = 0; v < 4; ++v) {
          af.u[v]     = pp[abase + v];
          af.u[4 + v] = pp[abase + 4 + v];
        }
        int bbase = (16 * cg + ln) * 33 + 8 * hg + 16 * ks;
#pragma unroll
        for (int v = 0; v < 8; ++v) bf.u[v] = vt32[bbase + v];
        accO[cg] = wmma_bf16(af, bf, accO[cg]);
      }
    }
    __syncthreads();   // protect Ks/VT before next stage
  }

  // epilogue: normalize, write O f32 in (T, B, E)
#pragma unroll
  for (int cg = 0; cg < 4; ++cg) {
    int d = 16 * cg + ln;
#pragma unroll
    for (int g = 0; g < 8; ++g) {
      int trow = t0 + 16 * w + g + 8 * hg;
      float val = accO[cg][g] / fmaxf(l_run[g], 1e-30f);
      O[((size_t)trow * BATCH + b) * EMB + h * HD + d] = val;
    }
  }
}

// ---------------------------------------------------------------------------
// Kernel 3: output projection  out = O @ Wo^T + bo   (f32 in/out, bf16 WMMA)
// ---------------------------------------------------------------------------
__global__ __launch_bounds__(128) void out_proj_kernel(
    const float* __restrict__ X, const float* __restrict__ Wo,
    const float* __restrict__ bo, float* __restrict__ out)
{
  __shared__ unsigned short Xs[64 * 32];
  __shared__ unsigned short Ws[64 * 32];

  const int tid  = threadIdx.x;
  const int lane = tid & 31;
  const int w    = tid >> 5;
  const int hg   = lane >> 4;
  const int ln   = lane & 15;
  const int n0 = blockIdx.x * 64;
  const int j0 = blockIdx.y * 64;

  v8f acc[4];
#pragma unroll
  for (int cg = 0; cg < 4; ++cg) acc[cg] = (v8f){0,0,0,0,0,0,0,0};

  for (int kk = 0; kk < EMB; kk += 32) {
    for (int idx = tid; idx < 64 * 32; idx += 128) {
      int r = idx >> 5, c = idx & 31;
      Xs[idx] = f2bf(X[(size_t)(n0 + r) * EMB + kk + c]);
      Ws[idx] = f2bf(Wo[(size_t)(j0 + r) * EMB + kk + c]);
    }
    __syncthreads();
    const unsigned int* xs32 = (const unsigned int*)Xs;
    const unsigned int* ws32 = (const unsigned int*)Ws;
    BFr a;
    {
      int base = (16 * w + ln) * 16 + 8 * hg;
#pragma unroll
      for (int v = 0; v < 4; ++v) {
        a.u[v]     = xs32[base + v];
        a.u[4 + v] = xs32[base + 4 + v];
      }
    }
#pragma unroll
    for (int cg = 0; cg < 4; ++cg) {
      BFr bfr;
      int jb = (16 * cg + ln) * 16 + 8 * hg;
#pragma unroll
      for (int v = 0; v < 8; ++v) bfr.u[v] = ws32[jb + v];
      acc[cg] = wmma_bf16(a, bfr, acc[cg]);
    }
    __syncthreads();
  }

#pragma unroll
  for (int cg = 0; cg < 4; ++cg) {
    int col = j0 + 16 * cg + ln;
    float bb = bo[col];
#pragma unroll
    for (int g = 0; g < 8; ++g) {
      int row = n0 + 16 * w + g + 8 * hg;
      out[(size_t)row * EMB + col] = acc[cg][g] + bb;
    }
  }
}

// ---------------------------------------------------------------------------
extern "C" void kernel_launch(void* const* d_in, const int* in_sizes, int n_in,
                              void* d_out, int out_size, void* d_ws, size_t ws_size,
                              hipStream_t stream) {
  const float*         query     = (const float*)d_in[0];
  const float*         attn_bias = (const float*)d_in[1];
  const unsigned char* kpm       = (const unsigned char*)d_in[2];
  const float*         Wq        = (const float*)d_in[3];
  const float*         bq        = (const float*)d_in[4];
  const float*         Wk        = (const float*)d_in[5];
  const float*         Wv        = (const float*)d_in[6];
  const float*         bv        = (const float*)d_in[7];
  const float*         Wo        = (const float*)d_in[8];
  const float*         bo        = (const float*)d_in[9];
  float*               out       = (float*)d_out;

  const size_t HSZ = (size_t)BATCH * HEADS * T_SEQ * HD;  // 4 Mi elements
  unsigned short* Qh = (unsigned short*)d_ws;              // 8 MB
  unsigned short* Kh = Qh + HSZ;                           // 8 MB
  unsigned short* Vh = Kh + HSZ;                           // 8 MB
  float*          O  = (float*)(Vh + HSZ);                 // 16 MB  (total 40 MB)

  dim3 g1(NROW / 64, EMB / 64, 3);
  qkv_proj_rope_kernel<<<g1, 128, 0, stream>>>(query, Wq, bq, Wk, Wv, bv, Qh, Kh, Vh);

  dim3 g2(T_SEQ / 64, HEADS, BATCH);
  flash_attn_kernel<<<g2, 128, 0, stream>>>(Qh, Kh, Vh, attn_bias, kpm, O);

  dim3 g3(NROW / 64, EMB / 64, 1);
  out_proj_kernel<<<g3, 128, 0, stream>>>(O, Wo, bo, out);
}
